// InfoGraph_37864431681885
// MI455X (gfx1250) — matrix operand order repaired and checked by reading, not compiled, not run on hardware
//
#include <hip/hip_runtime.h>
#include <hip/hip_bf16.h>

typedef __attribute__((ext_vector_type(16))) __bf16       v16bf;
typedef __attribute__((ext_vector_type(8)))  float        v8f;
typedef __attribute__((ext_vector_type(8)))  unsigned int v8u;

#define LDSP 17  // u32 (bf16-pair) stride per LDS row: 16 pairs + 1 pad (gcd(17,64)=1)

__device__ __forceinline__ unsigned short f2bf(float f) {
  unsigned u = __float_as_uint(f);
  unsigned r = u + 0x7FFFu + ((u >> 16) & 1u);   // round-to-nearest-even
  return (unsigned short)(r >> 16);
}
__device__ __forceinline__ unsigned pack_bf2(float lo, float hi) {
  return (unsigned)f2bf(lo) | ((unsigned)f2bf(hi) << 16);
}
__device__ __forceinline__ float bf2f(unsigned b16) {
  return __uint_as_float(b16 << 16);
}

// Raw LDS byte offset of a __shared__ pointer (ptrtoint of addrspace(3)).
__device__ __forceinline__ unsigned lds_addr(const void* p) {
  return (unsigned)(uintptr_t)(__attribute__((address_space(3))) const void*)p;
}
// CDNA5 async global->LDS copy (16B per lane), tracked by ASYNCcnt.
__device__ __forceinline__ void async_copy_b128(unsigned* dst_lds, const unsigned* src) {
  asm volatile("global_load_async_to_lds_b128 %0, %1, off"
               :: "v"(lds_addr(dst_lds)), "v"(src) : "memory");
}
__device__ __forceinline__ void wait_async() {
  asm volatile("s_wait_asynccnt 0x0" ::: "memory");
}

// ---------------------------------------------------------------------------
// Pre-swizzle weights W[K][N] f32 -> bf16 pairs in WMMA-B tile order:
// Wb[((nblk*(K/32) + kstep)*TN + n_in_blk)*16 + kpair]   (u32 = 2 bf16)
// ---------------------------------------------------------------------------
__global__ __launch_bounds__(256)
void ig_swizzle(const float* __restrict__ W, unsigned* __restrict__ Wb,
                int K, int N, int TN)
{
  int gid   = blockIdx.x * 256 + threadIdx.x;
  int total = (K >> 1) * N;
  if (gid >= total) return;
  int n   = gid % N;
  int kpg = gid / N;
  int k   = kpg << 1;
  int s   = k >> 5;
  int kp  = (k & 31) >> 1;
  int b   = n / TN;
  int nn  = n % TN;
  size_t out = ((size_t)(b * (K >> 5) + s) * TN + nn) * 16 + kp;
  Wb[out] = pack_bf2(W[(size_t)k * N + n], W[(size_t)(k + 1) * N + n]);
}

// ---------------------------------------------------------------------------
// GEMM1: H[M,N] = bf16( X[M,K] @ W[K,N] + bias ), bf16 WMMA, f32 accum.
// Fused BN batch statistics (column sum / sumsq) via LDS + global f32 atomics.
// Block tile 128x128, k-step 32, double-buffered LDS, async B copies.
// ---------------------------------------------------------------------------
__global__ __launch_bounds__(256)
void ig_gemm1(const float* __restrict__ X, const unsigned* __restrict__ Wb,
              const float* __restrict__ bias, unsigned short* __restrict__ H,
              float* __restrict__ gsum, float* __restrict__ gssq,
              int M, int N, int K)
{
  __shared__ unsigned As[2][128 * LDSP];
  __shared__ unsigned Bs[2][128 * LDSP];
  __shared__ float csum[128];
  __shared__ float cssq[128];

  const int tid  = threadIdx.x;
  const int lane = tid & 31;
  const int w    = tid >> 5;
  const int wm   = w & 3;
  const int wn   = w >> 2;
  const int half = lane >> 4;
  const int l16  = lane & 15;
  const int m0   = blockIdx.x * 128;
  const int n0   = blockIdx.y * 128;
  const int NT   = K >> 5;
  const bool full = (m0 + 128 <= M);

  if (tid < 128) { csum[tid] = 0.f; cssq[tid] = 0.f; }

  v8f acc[2][4];
  const v8f vzero = {0.f,0.f,0.f,0.f,0.f,0.f,0.f,0.f};
#pragma unroll
  for (int i = 0; i < 2; i++)
#pragma unroll
    for (int j = 0; j < 4; j++) acc[i][j] = vzero;

  const int qa = tid & 7;   // A staging: quad of 4 floats (2 bf16 pairs)
  const int ra = tid >> 3;  // A staging: row base, 32 rows per pass
  const unsigned* WbT = Wb + (size_t)blockIdx.y * NT * (128 * 16);

  auto stage = [&](int buf, int kt) {
    const int k0 = kt << 5;
    if (full) {
#pragma unroll
      for (int p = 0; p < 4; p++) {
        int r = ra + 32 * p;
        const float4 x = *(const float4*)(X + (size_t)(m0 + r) * K + k0 + 4 * qa);
        As[buf][r * LDSP + 2 * qa]     = pack_bf2(x.x, x.y);
        As[buf][r * LDSP + 2 * qa + 1] = pack_bf2(x.z, x.w);
      }
    } else {
#pragma unroll
      for (int p = 0; p < 4; p++) {
        int r = ra + 32 * p;
        float4 x = make_float4(0.f, 0.f, 0.f, 0.f);
        if (m0 + r < M)
          x = *(const float4*)(X + (size_t)(m0 + r) * K + k0 + 4 * qa);
        As[buf][r * LDSP + 2 * qa]     = pack_bf2(x.x, x.y);
        As[buf][r * LDSP + 2 * qa + 1] = pack_bf2(x.z, x.w);
      }
    }
    const unsigned* src = WbT + (size_t)kt * (128 * 16);
#pragma unroll
    for (int p = 0; p < 2; p++) {
      int ch = tid + 256 * p;
      int n  = ch >> 2, c4 = (ch & 3) * 4;
      async_copy_b128(&Bs[buf][n * LDSP + c4], src + ch * 4);
    }
  };

  stage(0, 0);
  for (int kt = 0; kt < NT; kt++) {
    wait_async();
    __syncthreads();
    if (kt + 1 < NT) stage((kt + 1) & 1, kt + 1);

    const unsigned* A = As[kt & 1];
    const unsigned* B = Bs[kt & 1];
    v8u afr[2], bfr[4];
#pragma unroll
    for (int i = 0; i < 2; i++) {
      int row = wm * 32 + i * 16 + l16;
#pragma unroll
      for (int v = 0; v < 8; v++) {
        int kp = v + 8 * half;
        afr[i][v] = A[row * LDSP + kp];
      }
    }
#pragma unroll
    for (int j = 0; j < 4; j++) {
      int col = wn * 64 + j * 16 + l16;
#pragma unroll
      for (int v = 0; v < 8; v++)
        bfr[j][v] = B[col * LDSP + v + 8 * half];
    }
#pragma unroll
    for (int i = 0; i < 2; i++)
#pragma unroll
      for (int j = 0; j < 4; j++)
        acc[i][j] = __builtin_amdgcn_wmma_f32_16x16x32_bf16(
            false, __builtin_bit_cast(v16bf, afr[i]),
            false, __builtin_bit_cast(v16bf, bfr[j]),
            (short)0, acc[i][j], false, false);
  }

  // --- epilogue: +bias, store bf16 H, fused column stats
#pragma unroll
  for (int i = 0; i < 2; i++)
#pragma unroll
    for (int j = 0; j < 4; j++) {
      float bcl = bias[n0 + wn * 64 + j * 16 + l16];
#pragma unroll
      for (int v = 0; v < 8; v++) acc[i][j][v] += bcl;
    }
#pragma unroll
  for (int i = 0; i < 2; i++)
#pragma unroll
    for (int j = 0; j < 4; j++) {
      int col = n0 + wn * 64 + j * 16 + l16;
#pragma unroll
      for (int v = 0; v < 8; v++) {
        int row = m0 + wm * 32 + i * 16 + half * 8 + v;
        if (row < M) H[(size_t)row * N + col] = f2bf(acc[i][j][v]);
      }
    }
#pragma unroll
  for (int j = 0; j < 4; j++) {
    float s = 0.f, ss = 0.f;
#pragma unroll
    for (int i = 0; i < 2; i++)
#pragma unroll
      for (int v = 0; v < 8; v++) {
        float h = acc[i][j][v];
        int row = m0 + wm * 32 + i * 16 + half * 8 + v;
        if (row < M) { s += h; ss += h * h; }
      }
    atomicAdd(&csum[wn * 64 + j * 16 + l16], s);
    atomicAdd(&cssq[wn * 64 + j * 16 + l16], ss);
  }
  __syncthreads();
  if (tid < 128) {
    atomicAdd(&gsum[n0 + tid], csum[tid]);
    atomicAdd(&gssq[n0 + tid], cssq[tid]);
  }
}

__global__ void ig_bnfin(const float* __restrict__ sum, const float* __restrict__ sumsq,
                         const float* __restrict__ g, const float* __restrict__ beta,
                         float invN, int N, float* __restrict__ a, float* __restrict__ c)
{
  int j = blockIdx.x * blockDim.x + threadIdx.x;
  if (j < N) {
    float mu  = sum[j] * invN;
    float var = sumsq[j] * invN - mu * mu;
    float aj  = g[j] * rsqrtf(var + 1e-5f);
    a[j] = aj;
    c[j] = beta[j] - mu * aj;
  }
}

// ---------------------------------------------------------------------------
// GEMM2: P[M,256] = l2norm_rows( relu(a*H + c) @ W2[1024,256] + b2 )
// Block tile 64x256 (full D_OUT per block), double-buffered, async B copies,
// BN+ReLU fused into A staging, bias + row L2-norm fused into epilogue.
// ---------------------------------------------------------------------------
__global__ __launch_bounds__(256)
void ig_gemm2(const unsigned short* __restrict__ H,
              const float* __restrict__ a, const float* __restrict__ c,
              const unsigned* __restrict__ Wb2, const float* __restrict__ b2,
              float* __restrict__ P, int M)
{
  constexpr int K = 1024;
  constexpr int N = 256;
  __shared__ unsigned As[2][64 * LDSP];
  __shared__ unsigned Bs[2][256 * LDSP];
  __shared__ float aS[K];
  __shared__ float cS[K];
  __shared__ float rowsq[64];

  const int tid  = threadIdx.x;
  const int lane = tid & 31;
  const int w    = tid >> 5;
  const int wm   = w & 1;
  const int wn   = w >> 1;
  const int half = lane >> 4;
  const int l16  = lane & 15;
  const int m0   = blockIdx.x * 64;
  const int NT   = K >> 5;
  const bool full = (m0 + 64 <= M);

  if (tid < 64) rowsq[tid] = 0.f;
#pragma unroll
  for (int p = 0; p < 4; p++) {
    aS[tid + 256 * p] = a[tid + 256 * p];
    cS[tid + 256 * p] = c[tid + 256 * p];
  }
  __syncthreads();   // aS/cS visible before first A staging

  v8f acc[2][4];
  const v8f vzero = {0.f,0.f,0.f,0.f,0.f,0.f,0.f,0.f};
#pragma unroll
  for (int i = 0; i < 2; i++)
#pragma unroll
    for (int j = 0; j < 4; j++) acc[i][j] = vzero;

  const int qa = tid & 3;   // A staging: 4 pairs (16B) per quad
  const int ra = tid >> 2;  // A staging: row (64 rows, one pass)

  auto stage = [&](int buf, int kt) {
    const int k0 = kt << 5;
    uint4 raw = make_uint4(0u, 0u, 0u, 0u);
    if (full || (m0 + ra < M))
      raw = *(const uint4*)(H + (size_t)(m0 + ra) * K + k0 + 8 * qa);
    const unsigned* rp = &raw.x;
#pragma unroll
    for (int i = 0; i < 4; i++) {
      int gk = k0 + 8 * qa + 2 * i;
      unsigned u = rp[i];
      float h0 = fmaxf(aS[gk]     * bf2f(u & 0xffffu) + cS[gk],     0.f);
      float h1 = fmaxf(aS[gk + 1] * bf2f(u >> 16)     + cS[gk + 1], 0.f);
      As[buf][ra * LDSP + 4 * qa + i] = pack_bf2(h0, h1);
    }
    const unsigned* src = Wb2 + (size_t)kt * (256 * 16);
#pragma unroll
    for (int p = 0; p < 4; p++) {
      int ch = tid + 256 * p;
      int n  = ch >> 2, c4 = (ch & 3) * 4;
      async_copy_b128(&Bs[buf][n * LDSP + c4], src + ch * 4);
    }
  };

  stage(0, 0);
  for (int kt = 0; kt < NT; kt++) {
    wait_async();
    __syncthreads();
    if (kt + 1 < NT) stage((kt + 1) & 1, kt + 1);

    const unsigned* A = As[kt & 1];
    const unsigned* B = Bs[kt & 1];
    v8u afr[2], bfr[4];
#pragma unroll
    for (int i = 0; i < 2; i++) {
      int row = wm * 32 + i * 16 + l16;
#pragma unroll
      for (int v = 0; v < 8; v++) {
        int kp = v + 8 * half;
        afr[i][v] = A[row * LDSP + kp];
      }
    }
#pragma unroll
    for (int j = 0; j < 4; j++) {
      int col = wn * 64 + j * 16 + l16;
#pragma unroll
      for (int v = 0; v < 8; v++)
        bfr[j][v] = B[col * LDSP + v + 8 * half];
    }
#pragma unroll
    for (int i = 0; i < 2; i++)
#pragma unroll
      for (int j = 0; j < 4; j++)
        acc[i][j] = __builtin_amdgcn_wmma_f32_16x16x32_bf16(
            false, __builtin_bit_cast(v16bf, afr[i]),
            false, __builtin_bit_cast(v16bf, bfr[j]),
            (short)0, acc[i][j], false, false);
  }

  // --- bias
#pragma unroll
  for (int i = 0; i < 2; i++)
#pragma unroll
    for (int j = 0; j < 4; j++) {
      float bcl = b2[wn * 64 + j * 16 + l16];
#pragma unroll
      for (int v = 0; v < 8; v++) acc[i][j][v] += bcl;
    }
  // --- row sum of squares: shfl-xor within 16-lane half, LDS atomics cross-wave
#pragma unroll
  for (int i = 0; i < 2; i++)
#pragma unroll
    for (int v = 0; v < 8; v++) {
      float p = 0.f;
#pragma unroll
      for (int j = 0; j < 4; j++) p += acc[i][j][v] * acc[i][j][v];
      p += __shfl_xor(p, 1); p += __shfl_xor(p, 2);
      p += __shfl_xor(p, 4); p += __shfl_xor(p, 8);
      if (l16 == 0) atomicAdd(&rowsq[wm * 32 + i * 16 + half * 8 + v], p);
    }
  __syncthreads();
  // --- scale and store
#pragma unroll
  for (int i = 0; i < 2; i++)
#pragma unroll
    for (int j = 0; j < 4; j++) {
      int col = wn * 64 + j * 16 + l16;
#pragma unroll
      for (int v = 0; v < 8; v++) {
        int r    = wm * 32 + i * 16 + half * 8 + v;
        int grow = m0 + r;
        if (grow < M) {
          float s = 1.f / fmaxf(sqrtf(rowsq[r]), 1e-12f);
          P[(size_t)grow * N + col] = acc[i][j][v] * s;
        }
      }
    }
}

// ---------------------------------------------------------------------------
__global__ __launch_bounds__(256)
void ig_wdg(const float* __restrict__ GP, const float* __restrict__ Wd,
            float* __restrict__ Wdg)
{
  int d = threadIdx.x;
  int g = blockIdx.x;
  const float* wr = Wd + (size_t)d * 256;
  const float* gr = GP + (size_t)g * 256;
  float s = 0.f;
#pragma unroll 8
  for (int e = 0; e < 256; e++) s += wr[e] * gr[e];
  Wdg[(size_t)g * 256 + d] = s;
}

__global__ __launch_bounds__(256)
void ig_score(const float* __restrict__ NP, const float* __restrict__ Wdg,
              const int* __restrict__ batch, const int* __restrict__ negr,
              const float* __restrict__ bdp, int Nn, float* __restrict__ accum)
{
  int gw   = (int)((blockIdx.x * blockDim.x + threadIdx.x) >> 5);
  int nwav = (int)((gridDim.x * blockDim.x) >> 5);
  int lane = threadIdx.x & 31;
  float bd = bdp[0];
  float pos_s = 0.f, neg_s = 0.f;
  for (int n = gw; n < Nn; n += nwav) {
    int b  = batch[n];
    int nr = negr[n];
    int ni = nr + (nr >= b ? 1 : 0);
    const float* np = NP  + (size_t)n  * 256 + lane * 8;
    const float* wp = Wdg + (size_t)b  * 256 + lane * 8;
    const float* wq = Wdg + (size_t)ni * 256 + lane * 8;
    float p = 0.f, q = 0.f;
#pragma unroll
    for (int e = 0; e < 8; e++) { float x = np[e]; p += x * wp[e]; q += x * wq[e]; }
#pragma unroll
    for (int m = 16; m >= 1; m >>= 1) { p += __shfl_xor(p, m); q += __shfl_xor(q, m); }
    if (lane == 0) {
      float sp = p + bd, sn = q + bd;
      pos_s += fmaxf(-sp, 0.f) + log1pf(expf(-fabsf(sp)));  // softplus(-sp)
      neg_s += fmaxf( sn, 0.f) + log1pf(expf(-fabsf(sn)));  // softplus(sn)
    }
  }
  if (lane == 0) { atomicAdd(&accum[0], pos_s); atomicAdd(&accum[1], neg_s); }
}

__global__ void ig_fin(const float* __restrict__ accum, float invN,
                       float* __restrict__ out)
{
  if (threadIdx.x == 0) out[0] = 0.5f * (accum[0] * invN + accum[1] * invN);
}

// ---------------------------------------------------------------------------
extern "C" void kernel_launch(void* const* d_in, const int* in_sizes, int n_in,
                              void* d_out, int out_size, void* d_ws, size_t ws_size,
                              hipStream_t stream)
{
  constexpr int NN = 100000, NG = 1024, DI = 512, DH = 1024, DO = 256;

  const float* x_node = (const float*)d_in[0];
  const float* x_graph= (const float*)d_in[1];
  const int*   batch  = (const int*)  d_in[2];
  const int*   negr   = (const int*)  d_in[3];
  const float* Wn1    = (const float*)d_in[4];
  const float* bn1    = (const float*)d_in[5];
  const float* gn1    = (const float*)d_in[6];
  const float* betan1 = (const float*)d_in[7];
  const float* Wn2    = (const float*)d_in[8];
  const float* bn2    = (const float*)d_in[9];
  const float* Wg1    = (const float*)d_in[10];
  const float* bg1    = (const float*)d_in[11];
  const float* gg1    = (const float*)d_in[12];
  const float* betag1 = (const float*)d_in[13];
  const float* Wg2    = (const float*)d_in[14];
  const float* bg2    = (const float*)d_in[15];
  const float* Wd     = (const float*)d_in[16];
  const float* bd     = (const float*)d_in[17];
  float* out = (float*)d_out;

  size_t off = 0;
  auto alloc = [&](size_t bytes) {
    void* p = (char*)d_ws + off;
    off += (bytes + 255) & ~(size_t)255;
    return p;
  };
  unsigned short* Hn = (unsigned short*)alloc((size_t)NN * DH * 2);
  unsigned short* Hg = (unsigned short*)alloc((size_t)NG * DH * 2);
  float* Pn    = (float*)alloc((size_t)NN * DO * 4);
  float* Pg    = (float*)alloc((size_t)NG * DO * 4);
  float* Wdg   = (float*)alloc((size_t)NG * DO * 4);
  unsigned* Wbn1 = (unsigned*)alloc((size_t)(DI / 2) * DH * 4);
  unsigned* Wbg1 = (unsigned*)alloc((size_t)(DI / 2) * DH * 4);
  unsigned* Wbn2 = (unsigned*)alloc((size_t)(DH / 2) * DO * 4);
  unsigned* Wbg2 = (unsigned*)alloc((size_t)(DH / 2) * DO * 4);
  float* sum_n = (float*)alloc(DH * 4);
  float* ssq_n = (float*)alloc(DH * 4);
  float* a_n   = (float*)alloc(DH * 4);
  float* c_n   = (float*)alloc(DH * 4);
  float* sum_g = (float*)alloc(DH * 4);
  float* ssq_g = (float*)alloc(DH * 4);
  float* a_g   = (float*)alloc(DH * 4);
  float* c_g   = (float*)alloc(DH * 4);
  float* accum = (float*)alloc(2 * 4);

  hipMemsetAsync(sum_n, 0, DH * 4, stream);
  hipMemsetAsync(ssq_n, 0, DH * 4, stream);
  hipMemsetAsync(sum_g, 0, DH * 4, stream);
  hipMemsetAsync(ssq_g, 0, DH * 4, stream);
  hipMemsetAsync(accum, 0, 2 * 4, stream);

  // --- weight pre-swizzle to WMMA-B bf16 tile layout
  ig_swizzle<<<(DI / 2) * DH / 256, 256, 0, stream>>>(Wn1, Wbn1, DI, DH, 128);
  ig_swizzle<<<(DI / 2) * DH / 256, 256, 0, stream>>>(Wg1, Wbg1, DI, DH, 128);
  ig_swizzle<<<(DH / 2) * DO / 256, 256, 0, stream>>>(Wn2, Wbn2, DH, DO, 256);
  ig_swizzle<<<(DH / 2) * DO / 256, 256, 0, stream>>>(Wg2, Wbg2, DH, DO, 256);

  // --- GEMM1 (Linear1) with fused BN stats
  ig_gemm1<<<dim3((NN + 127) / 128, DH / 128), 256, 0, stream>>>(
      x_node, Wbn1, bn1, Hn, sum_n, ssq_n, NN, DH, DI);
  ig_gemm1<<<dim3((NG + 127) / 128, DH / 128), 256, 0, stream>>>(
      x_graph, Wbg1, bg1, Hg, sum_g, ssq_g, NG, DH, DI);

  ig_bnfin<<<DH / 256, 256, 0, stream>>>(sum_n, ssq_n, gn1, betan1,
                                         1.f / (float)NN, DH, a_n, c_n);
  ig_bnfin<<<DH / 256, 256, 0, stream>>>(sum_g, ssq_g, gg1, betag1,
                                         1.f / (float)NG, DH, a_g, c_g);

  // --- BN+ReLU -> Linear2 -> l2norm (fused)
  ig_gemm2<<<(NN + 63) / 64, 256, 0, stream>>>(Hn, a_n, c_n, Wbn2, bn2, Pn, NN);
  ig_gemm2<<<(NG + 63) / 64, 256, 0, stream>>>(Hg, a_g, c_g, Wbg2, bg2, Pg, NG);

  // --- Wd @ graph_proj^T table, then scoring + loss
  ig_wdg<<<NG, 256, 0, stream>>>(Pg, Wd, Wdg);
  ig_score<<<512, 256, 0, stream>>>(Pn, Wdg, batch, negr, bd, NN, accum);
  ig_fin<<<1, 32, 0, stream>>>(accum, 1.f / (float)NN, out);
}
